// RecurrentGCN_44160853737699
// MI455X (gfx1250) — matrix operand order repaired and checked by reading, not compiled, not run on hardware
//
#include <hip/hip_runtime.h>

#define FEAT 64
#define HIDN 32
#define NCLS 10
#define TPW  4          // tiles (of 16 nodes) per wave

typedef __attribute__((ext_vector_type(16))) __bf16        v16bf;
typedef __attribute__((ext_vector_type(8)))  float         v8f;
typedef __attribute__((ext_vector_type(8)))  unsigned int  v8u;
typedef __attribute__((ext_vector_type(2)))  float         f2v;

// float -> bf16 (RTNE) in low 16 bits -- only used in the cold prep kernel
static __device__ __forceinline__ unsigned int f2bf(float f) {
  union { float f; unsigned int u; } v; v.f = f;
  unsigned int u = v.u;
  u += 0x7FFFu + ((u >> 16) & 1u);
  return u >> 16;
}
static __device__ __forceinline__ unsigned int packbf(float lo, float hi) {
  return f2bf(lo) | (f2bf(hi) << 16);
}
// reinterpret 8 packed dwords as a bf16x16 fragment (no conversion)
static __device__ __forceinline__ v16bf frag_bf(const unsigned int* p) {
  union { v8u u; v16bf b; } c;
#pragma unroll
  for (int i = 0; i < 8; ++i) c.u[i] = p[i];
  return c.b;
}

// ---------------------------------------------------------------------------
// Prep: fold (W[0]+W[1])[:64,:] for z/h gates and pad Wl (32x10 -> 32x16),
// emit bf16 B-operand fragments in the ISA wave32 register-image layout:
//   lane l, vgpr v: n = l%16, k = 32*kchunk + 8*(l/16) + 2*v + (v>=4 ? 8 : 0)  (pair k,k+1)
// ws (u32): [0..1023] Bz frags (t*2+c)*256+v*32+lane, [1024..2047] Bh,
//           [2048..2303] Bl.
// ---------------------------------------------------------------------------
__global__ void wmma_prep_weights(const float* __restrict__ Wz,
                                  const float* __restrict__ Wh,
                                  const float* __restrict__ Wl,
                                  unsigned int* __restrict__ ws) {
  const int lane = threadIdx.x & 31;
  const int n16  = lane & 15;
  const int kh   = lane >> 4;
#pragma unroll
  for (int g = 0; g < 2; ++g) {
    const float* W = (g == 0) ? Wz : Wh;   // shape (2, 96, 32); only k<64 used
#pragma unroll
    for (int t = 0; t < 2; ++t)
#pragma unroll
      for (int c = 0; c < 2; ++c)
#pragma unroll
        for (int v = 0; v < 8; ++v) {
          const int k0 = 32 * c + 8 * kh + 2 * v + ((v >= 4) ? 8 : 0);
          const int n  = 16 * t + n16;
          const float a0 = W[k0 * HIDN + n]       + W[3072 + k0 * HIDN + n];
          const float a1 = W[(k0 + 1) * HIDN + n] + W[3072 + (k0 + 1) * HIDN + n];
          ws[g * 1024 + (t * 2 + c) * 256 + v * 32 + lane] = packbf(a0, a1);
        }
  }
#pragma unroll
  for (int v = 0; v < 8; ++v) {             // Wl: (32, 10), pad cols 10..15 = 0
    const int k0 = 8 * kh + 2 * v + ((v >= 4) ? 8 : 0);
    const float a0 = (n16 < NCLS) ? Wl[k0 * NCLS + n16] : 0.0f;
    const float a1 = (n16 < NCLS) ? Wl[(k0 + 1) * NCLS + n16] : 0.0f;
    ws[2048 + v * 32 + lane] = packbf(a0, a1);
  }
}

// ---------------------------------------------------------------------------
// Fused per-node GRU-collapse + classifier. One wave handles TPW 16-node
// tiles so the 9 weight fragments are loaded once per wave.
// ---------------------------------------------------------------------------
__global__ __launch_bounds__(256) void rgcn_fused_wmma(
    const float* __restrict__ x,
    const float* __restrict__ bz, const float* __restrict__ bh,
    const float* __restrict__ bl,
    const unsigned int* __restrict__ ws,
    float* __restrict__ out, int numTiles)
{
  __shared__ __bf16 hst[8][16 * HIDN];     // per-wave 16x32 bf16 staging tile

  const int lane = threadIdx.x & 31;
  const int wv   = threadIdx.x >> 5;
  const int n16  = lane & 15;
  const int half = lane >> 4;

  // ---- B fragments: loaded once, register-resident for all TPW tiles ----
  v16bf zB[2][2], hB[2][2], lBf;
#pragma unroll
  for (int t = 0; t < 2; ++t)
#pragma unroll
    for (int c = 0; c < 2; ++c) {
      unsigned int tz[8], th[8];
#pragma unroll
      for (int v = 0; v < 8; ++v) {
        tz[v] = ws[(t * 2 + c) * 256 + v * 32 + lane];
        th[v] = ws[1024 + (t * 2 + c) * 256 + v * 32 + lane];
      }
      zB[t][c] = frag_bf(tz);
      hB[t][c] = frag_bf(th);
    }
  {
    unsigned int tl[8];
#pragma unroll
    for (int v = 0; v < 8; ++v) tl[v] = ws[2048 + v * 32 + lane];
    lBf = frag_bf(tl);
  }

  // ---- per-column biases (uniform across the tile loop) ----
  float bzv[2], bhv[2];
#pragma unroll
  for (int t = 0; t < 2; ++t) {
    bzv[t] = bz[16 * t + n16];
    bhv[t] = bh[16 * t + n16];
  }
  const float blv = (n16 < NCLS) ? bl[n16] : 0.0f;

  int tile = (blockIdx.x * 8 + wv) * TPW;
#pragma unroll 1
  for (int it = 0; it < TPW; ++it, ++tile) {
    if (tile >= numTiles) break;           // wave-uniform exit

    // ---- A fragments: 16x64 bf16 tile of x (ISA A layout) ----
    // m = lane%16; vgpr v<4 -> k = 8*half + 2v ; v>=4 -> k = 16+8*half+2(v-4)
    const float* xr = x + (long)(tile * 16 + n16) * FEAT;
    v16bf A[2];
#pragma unroll
    for (int c = 0; c < 2; ++c)
#pragma unroll
      for (int v = 0; v < 8; ++v) {
        const int kk = 2 * v + ((v >= 4) ? 8 : 0);
        const int k  = 32 * c + 8 * half + kk;
        const f2v p = __builtin_nontemporal_load((const f2v*)(xr + k)); // 8B aligned
        A[c][2 * v]     = (__bf16)p.x;
        A[c][2 * v + 1] = (__bf16)p.y;
      }

    // ---- gate GEMMs ----
    v8f cz[2], ch[2];
#pragma unroll
    for (int t = 0; t < 2; ++t) {
      v8f az, ah;
#pragma unroll
      for (int r = 0; r < 8; ++r) { az[r] = 0.0f; ah[r] = 0.0f; }
      az = __builtin_amdgcn_wmma_f32_16x16x32_bf16(false, A[0], false, zB[t][0], (short)0, az, false, false);
      az = __builtin_amdgcn_wmma_f32_16x16x32_bf16(false, A[1], false, zB[t][1], (short)0, az, false, false);
      ah = __builtin_amdgcn_wmma_f32_16x16x32_bf16(false, A[0], false, hB[t][0], (short)0, ah, false, false);
      ah = __builtin_amdgcn_wmma_f32_16x16x32_bf16(false, A[1], false, hB[t][1], (short)0, ah, false, false);
      cz[t] = az; ch[t] = ah;
    }

    // ---- activations on C fragments (m = r + 8*half, n = 16t + n16) ----
    // branch-free fast math (hardware exp/rcp):
    //   1 - sigmoid(a) = e / (1 + e),          e = exp(-a)
    //   tanh(b)        = 1 - 2 / (exp(2b) + 1)   (saturates correctly at +/-1)
#pragma unroll
    for (int t = 0; t < 2; ++t) {
      const int n = 16 * t + n16;
#pragma unroll
      for (int r = 0; r < 8; ++r) {
        const float a   = cz[t][r] + bzv[t];
        const float b   = ch[t][r] + bhv[t];
        const float ez  = __expf(-a);
        const float omz = ez * __builtin_amdgcn_rcpf(1.0f + ez);     // 1 - Z
        const float e2b = __expf(2.0f * b);
        const float th  = fmaf(-2.0f, __builtin_amdgcn_rcpf(1.0f + e2b), 1.0f); // tanh
        const float h   = fmaxf(omz * th, 0.0f);                     // relu(H)
        hst[wv][(r + 8 * half) * HIDN + n] = (__bf16)h;
      }
    }

    // ---- reload staged tile as classifier A fragment (same-wave LDS) ----
    v16bf Ah;
#pragma unroll
    for (int v = 0; v < 8; ++v) {
      const int kk  = 2 * v + ((v >= 4) ? 8 : 0);
      const int idx = n16 * HIDN + 8 * half + kk;
      Ah[2 * v]     = hst[wv][idx];
      Ah[2 * v + 1] = hst[wv][idx + 1];
    }

    // ---- classifier: acc = bl + relu(H) @ Wl ----
    v8f acc;
#pragma unroll
    for (int r = 0; r < 8; ++r) acc[r] = blv;
    acc = __builtin_amdgcn_wmma_f32_16x16x32_bf16(false, Ah, false, lBf, (short)0, acc, false, false);

    if (n16 < NCLS) {
      const long base = (long)tile * 16 * NCLS;
#pragma unroll
      for (int r = 0; r < 8; ++r)
        __builtin_nontemporal_store(acc[r], out + base + (long)(r + 8 * half) * NCLS + n16);
    }
  }
}

extern "C" void kernel_launch(void* const* d_in, const int* in_sizes, int n_in,
                              void* d_out, int out_size, void* d_ws, size_t ws_size,
                              hipStream_t stream) {
  // setup_inputs order: x, edge_index, edge_weight, Wz, bz, Wr, br, Wh, bh, Wl, bl
  const float* x  = (const float*)d_in[0];
  const float* Wz = (const float*)d_in[3];
  const float* bz = (const float*)d_in[4];
  const float* Wh = (const float*)d_in[7];
  const float* bh = (const float*)d_in[8];
  const float* Wl = (const float*)d_in[9];
  const float* bl = (const float*)d_in[10];
  float* out = (float*)d_out;
  unsigned int* ws = (unsigned int*)d_ws;   // uses 2304 u32 = 9216 bytes

  const int N        = in_sizes[0] / FEAT;        // 1,000,000
  const int numTiles = N / 16;                    // 62,500
  const int blocks   = (numTiles + 8 * TPW - 1) / (8 * TPW);

  hipLaunchKernelGGL(wmma_prep_weights, dim3(1), dim3(32), 0, stream, Wz, Wh, Wl, ws);
  hipLaunchKernelGGL(rgcn_fused_wmma, dim3(blocks), dim3(256), 0, stream,
                     x, bz, bh, bl, (const unsigned int*)ws, out, numTiles);
}
